// TemporalMultiHeadAttention_84902913508043
// MI455X (gfx1250) — matrix operand order repaired and checked by reading, not compiled, not run on hardware
//
#include <hip/hip_runtime.h>
#include <hip/hip_bf16.h>
#include <math.h>

// MI455X / gfx1250, wave32. Fused edge-biased MHA.
// B=4 N=512 D=128 H=8 HD=16. h_edges (512 MiB fp32) dominates: memory-bound,
// single streaming pass + online softmax. Matrix ops in bf16 WMMA (f32 acc).
// Edge stream staged into per-wave double-buffered LDS tiles using BOTH
// CDNA5 async paths: TENSOR_LOAD_TO_LDS (TDM, TENSORcnt) for even tiles and
// GLOBAL_LOAD_ASYNC_TO_LDS_B128 (ASYNCcnt) for odd tiles.

#define B_  4
#define N_  512
#define D_  128
#define H_  8

typedef __attribute__((ext_vector_type(16))) __bf16   v16bf;
typedef __attribute__((ext_vector_type(8)))  __bf16   v8bf;
typedef __attribute__((ext_vector_type(8)))  float    v8f;
typedef __attribute__((ext_vector_type(4)))  uint32_t u32x4;
typedef __attribute__((ext_vector_type(8)))  uint32_t u32x8;

// ---- WMMA wrapper: D = A(16x32 bf16) * B(32x16 bf16) + C(16x16 f32) ----
__device__ __forceinline__ v8f wmma_bf16(v16bf a, v16bf b, v8f c) {
    return __builtin_amdgcn_wmma_f32_16x16x32_bf16(
        /*neg_a=*/false, a, /*neg_b=*/false, b,
        /*c_mod=*/(short)0, c, /*reuse_a=*/false, /*reuse_b=*/false);
}

// ---- CDNA5 TDM: one-shot DMA of an 8 KiB contiguous tile to LDS ----
// D# group0: count=1 | lds_addr | global_addr[56:0] | type=2 ("image")
// D# group1: data_size=4B, 1-D tile: tensor_dim0 = tile_dim0 = 2048 elems.
__device__ __forceinline__ void tdm_load_8k(uint32_t lds_byte, uint64_t gaddr) {
    u32x4 g0;
    g0[0] = 1u;                                                    // count=1
    g0[1] = lds_byte;                                              // lds_addr
    g0[2] = (uint32_t)gaddr;                                       // gaddr[31:0]
    g0[3] = ((uint32_t)(gaddr >> 32) & 0x01FFFFFFu) | (2u << 30);  // gaddr[56:32]|type=2
    u32x8 g1;
    g1[0] = (2u << 16);          // workgroup_mask=0, data_size=2 (4B), no flags
    g1[1] = (2048u & 0xFFFFu) << 16;   // tensor_dim0[15:0] = 2048
    g1[2] = (1u << 16);          // tensor_dim0[31:16]=0 | tensor_dim1=1
    g1[3] = (2048u << 16);       // tensor_dim1[31:16]=0 | tile_dim0=2048
    g1[4] = 0u;                  // tile_dim1=0 (1-D), tile_dim2=0
    g1[5] = 2048u;               // tensor_dim0_stride[31:0]
    g1[6] = 0u;
    g1[7] = 0u;
    asm volatile("tensor_load_to_lds %0, %1" :: "s"(g0), "s"(g1) : "memory");
}

// ---- CDNA5 async copy: 8 KiB contiguous global -> LDS (per wave) ----
// 16 chunks x 32 lanes x 16B, tracked by ASYNCcnt.
__device__ __forceinline__ void stage_tile_async(uint32_t lds_byte, uint64_t gaddr) {
#pragma unroll
    for (int kc = 0; kc < 16; ++kc) {
        asm volatile("global_load_async_to_lds_b128 %0, %1, off"
                     :: "v"(lds_byte), "v"(gaddr) : "memory");
        lds_byte += 512;
        gaddr    += 512;
    }
}
__device__ __forceinline__ void wait_tensor0() {
    asm volatile("s_wait_tensorcnt 0x0" ::: "memory");
}
__device__ __forceinline__ void wait_async0() {
    asm volatile("s_wait_asynccnt 0x0" ::: "memory");
}
__device__ __forceinline__ void wait_ds0() {   // intra-wave LDS exchange fence
    asm volatile("s_wait_dscnt 0x0" ::: "memory");
}

// ---- fragment builders (wave32; lane = half*16 + c) ----
// A 16x32 bf16 layout: lane(half,c): row m=c, elems 0..15 = K=kb+half*16+t.
__device__ __forceinline__ void pack8(v16bf& a, int base, float4 x, float4 y) {
    a[base+0]=(__bf16)x.x; a[base+1]=(__bf16)x.y; a[base+2]=(__bf16)x.z; a[base+3]=(__bf16)x.w;
    a[base+4]=(__bf16)y.x; a[base+5]=(__bf16)y.y; a[base+6]=(__bf16)y.z; a[base+7]=(__bf16)y.w;
}

__device__ __forceinline__ v16bf a_frag_f32(const float* __restrict__ row, int kb, int half) {
    const float4* p = (const float4*)(row + kb + half*16);
    v16bf a;
    pack8(a, 0, p[0], p[1]);   // K = kb+half*16 .. +7
    pack8(a, 8, p[2], p[3]);   // K = kb+half*16+8 .. +15
    return a;
}

__device__ __forceinline__ v16bf a_frag_bf16(const __bf16* __restrict__ row, int kb, int half) {
    const v8bf* p = (const v8bf*)(row + kb + half*16);
    v8bf lo = p[0], hi = p[1];
    v16bf a;
#pragma unroll
    for (int t = 0; t < 8; ++t) { a[t] = lo[t]; a[t+8] = hi[t]; }
    return a;
}

// B 32x16 bf16 layout: lane(half,c): col n=c, elems 0..15 = K=kb+half*16+t.
// Bt[n][k] == row-major weight row n -> load 16 consecutive floats.
__device__ __forceinline__ v16bf b_frag_f32(const float* __restrict__ row, int kb, int half) {
    const float4* p = (const float4*)(row + kb + half*16);
    v16bf b;
    pack8(b, 0, p[0], p[1]);
    pack8(b, 8, p[2], p[3]);
    return b;
}

// =====================================================================
// Kernel 1: q,k,v = h_nodes @ W{q,k,v}.T   (q,v -> f32 ws; k -> bf16 ws)
// =====================================================================
__global__ __launch_bounds__(256) void qkv_kernel(
    const float* __restrict__ hn,
    const float* __restrict__ Wq, const float* __restrict__ Wk,
    const float* __restrict__ Wv,
    float* __restrict__ q_ws, __bf16* __restrict__ k_ws,
    float* __restrict__ v_ws) {
    const int b = blockIdx.y, i0 = blockIdx.x * 16;
    const int w = threadIdx.x >> 5, lane = threadIdx.x & 31;
    const int half = lane >> 4, c = lane & 15;

    const float* arow = hn + ((size_t)(b * N_) + i0 + c) * D_;
    v16bf A[4];
#pragma unroll
    for (int s = 0; s < 4; ++s) A[s] = a_frag_f32(arow, s * 32, half);

    const size_t wr = (size_t)(w * 16 + c) * D_;
    v8f dq = {}, dk = {}, dv = {};
#pragma unroll
    for (int s = 0; s < 4; ++s) {
        dq = wmma_bf16(A[s], b_frag_f32(Wq + wr, s * 32, half), dq);
        dk = wmma_bf16(A[s], b_frag_f32(Wk + wr, s * 32, half), dk);
        dv = wmma_bf16(A[s], b_frag_f32(Wv + wr, s * 32, half), dv);
    }
#pragma unroll
    for (int r = 0; r < 8; ++r) {
        size_t idx = ((size_t)(b * N_) + i0 + half * 8 + r) * D_ + w * 16 + c;
        q_ws[idx] = dq[r];
        k_ws[idx] = (__bf16)dk[r];
        v_ws[idx] = dv[r];
    }
}

// =====================================================================
// Kernel 2: fused edge-bias attention, one (b,i) per block, 8 waves.
// Wave w streams j-tiles {w,w+8,w+16,w+24}: even tiles staged by TDM into
// buf0, odd tiles by async-to-LDS into buf1 (two independent DMA pipelines).
// =====================================================================
__global__ __launch_bounds__(256) void attn_kernel(
    const float* __restrict__ h_edges,
    const float* __restrict__ Web, const float* __restrict__ Weg,
    const float* __restrict__ q_ws, const __bf16* __restrict__ k_ws,
    const float* __restrict__ v_ws, float* __restrict__ attn_ws) {
    const int i = blockIdx.x, b = blockIdx.y;
    const int w = threadIdx.x >> 5, lane = threadIdx.x & 31;
    const int half = lane >> 4, c = lane & 15;

    __shared__ alignas(16) float ebuf[8][2][16 * D_];  // 128 KiB staging
    __shared__ float lds_q[D_];
    __shared__ float s_w[8][16][H_];   // [wave][j][h] gated exp weights
    __shared__ float s_r[8][H_];       // per-tile rescale factor
    __shared__ float s_m[8][H_], s_l[8][H_];
    __shared__ float s_acc[8][D_];

    if (threadIdx.x < D_)
        lds_q[threadIdx.x] = q_ws[((size_t)(b * N_) + i) * D_ + threadIdx.x];
    __syncthreads();

    // Hoisted B fragments.
    // Bbg: cols 0..7 = W_edge_bias rows, cols 8..15 = W_edge_gate rows.
    // Bq : block-diagonal q -> per-head scores in one K=128 chain.
    const float* wbg_row = (c < 8) ? (Web + (size_t)c * D_)
                                   : (Weg + (size_t)(c - 8) * D_);
    v16bf Bbg[4], Bq[4];
#pragma unroll
    for (int s = 0; s < 4; ++s) {
        Bbg[s] = b_frag_f32(wbg_row, s * 32, half);
        const int hk = s * 2 + half;             // head of K range [kb+half*16, +16)
        v16bf bq;
#pragma unroll
        for (int t = 0; t < 16; ++t)
            bq[t] = (c == hk) ? (__bf16)lds_q[s * 32 + half * 16 + t] : (__bf16)0.0f;
        Bq[s] = bq;
    }

    // Per-wave double buffer over the contiguous 8 KiB j-tiles.
    const uint64_t gbase = (uint64_t)(uintptr_t)(h_edges + ((size_t)b * N_ + i) * N_ * D_);
    const uint32_t lb0 = (uint32_t)(uintptr_t)&ebuf[w][0][0];  // TDM buffer
    const uint32_t lb1 = (uint32_t)(uintptr_t)&ebuf[w][1][0];  // async buffer

    tdm_load_8k(lb0, gbase + (size_t)(w * 16) * 512);          // tile 0 -> buf0

    float run_m = -1e30f, run_l = 0.0f;
    float acc0 = 0.f, acc1 = 0.f, acc2 = 0.f, acc3 = 0.f;

    for (int tt = 0; tt < 4; ++tt) {
        const int j0 = (w + tt * 8) * 16;
        if (tt < 3) {  // prefetch next tile into the other buffer
            const uint64_t ng = gbase + (size_t)((w + (tt + 1) * 8) * 16) * 512;
            if (tt & 1) tdm_load_8k(lb0, ng);                       // even tiles: TDM
            else        stage_tile_async(lb1 + lane * 16, ng + lane * 16); // odd: async
        }
        if (tt & 1) wait_async0();    // buf1 landed (ASYNCcnt)
        else        wait_tensor0();   // buf0 landed (TENSORcnt)

        const float*  etile = ebuf[w][tt & 1] + (size_t)c * D_;  // row m=c
        const __bf16* krow  = k_ws + ((size_t)(b * N_) + j0 + c) * D_;

        v16bf Ae[4], Ak[4];
#pragma unroll
        for (int s = 0; s < 4; ++s) Ak[s] = a_frag_bf16(krow, s * 32, half);
#pragma unroll
        for (int s = 0; s < 4; ++s) Ae[s] = a_frag_f32(etile, s * 32, half);

        v8f Dbg = {}, Ds = {};
#pragma unroll
        for (int s = 0; s < 4; ++s) {
            Dbg = wmma_bf16(Ae[s], Bbg[s], Dbg);
            Ds  = wmma_bf16(Ak[s], Bq[s], Ds);
        }

        // lanes c<8 own scores for head c, rows j = half*8 + r
        float g[8], sc[8];
#pragma unroll
        for (int r = 0; r < 8; ++r) {
            float gl = __shfl(Dbg[r], (lane + 8) & 31, 32);  // gate col = h+8
            g[r]  = 1.0f / (1.0f + __expf(-gl));
            sc[r] = Ds[r] * 0.25f + Dbg[r];                  // /sqrt(16) + bias
        }
        float tm = sc[0];
#pragma unroll
        for (int r = 1; r < 8; ++r) tm = fmaxf(tm, sc[r]);
        tm = fmaxf(tm, __shfl_xor(tm, 16, 32));              // combine halves
        const float nm = fmaxf(run_m, tm);
        const float rs = __expf(run_m - nm);
        float p[8], tsum = 0.f;
#pragma unroll
        for (int r = 0; r < 8; ++r) { p[r] = __expf(sc[r] - nm); tsum += p[r]; }
        tsum += __shfl_xor(tsum, 16, 32);
        run_l = run_l * rs + tsum;
        run_m = nm;

        if (c < H_) {
#pragma unroll
            for (int r = 0; r < 8; ++r) s_w[w][half * 8 + r][c] = p[r] * g[r];
            if (half == 0) s_r[w][c] = rs;
        }
        wait_ds0();  // intra-wave exchange only; LDS is in-order per wave

        // V accumulation: lane owns dims d = lane+32t, head(d) = half+2t
        const float r0 = s_r[w][half + 0], r1 = s_r[w][half + 2];
        const float r2 = s_r[w][half + 4], r3 = s_r[w][half + 6];
        acc0 *= r0; acc1 *= r1; acc2 *= r2; acc3 *= r3;
        const float* vbase = v_ws + ((size_t)(b * N_) + j0) * D_ + lane;
#pragma unroll
        for (int j = 0; j < 16; ++j) {
            const float* vr = vbase + (size_t)j * D_;
            acc0 += s_w[w][j][half + 0] * vr[0];
            acc1 += s_w[w][j][half + 2] * vr[32];
            acc2 += s_w[w][j][half + 4] * vr[64];
            acc3 += s_w[w][j][half + 6] * vr[96];
        }
    }

    // Flash merge of 8 per-wave partial states.
    if (c < H_ && half == 0) { s_m[w][c] = run_m; s_l[w][c] = run_l; }
    s_acc[w][lane]      = acc0;
    s_acc[w][lane + 32] = acc1;
    s_acc[w][lane + 64] = acc2;
    s_acc[w][lane + 96] = acc3;
    __syncthreads();
    if (threadIdx.x < D_) {
        const int d = threadIdx.x, h = d >> 4;
        float M = -1e30f;
#pragma unroll
        for (int ww = 0; ww < 8; ++ww) M = fmaxf(M, s_m[ww][h]);
        float L = 0.f, O = 0.f;
#pragma unroll
        for (int ww = 0; ww < 8; ++ww) {
            const float e = __expf(s_m[ww][h] - M);
            L += s_l[ww][h] * e;
            O += s_acc[ww][d] * e;
        }
        attn_ws[((size_t)(b * N_) + i) * D_ + d] = O / L;
    }
}

// =====================================================================
// Kernel 3: out = attn_ws @ Wo.T
// =====================================================================
__global__ __launch_bounds__(256) void out_kernel(
    const float* __restrict__ attn_ws, const float* __restrict__ Wo,
    float* __restrict__ out) {
    const int b = blockIdx.y, i0 = blockIdx.x * 16;
    const int w = threadIdx.x >> 5, lane = threadIdx.x & 31;
    const int half = lane >> 4, c = lane & 15;

    const float* arow = attn_ws + ((size_t)(b * N_) + i0 + c) * D_;
    const float* wrow = Wo + (size_t)(w * 16 + c) * D_;
    v8f d = {};
#pragma unroll
    for (int s = 0; s < 4; ++s)
        d = wmma_bf16(a_frag_f32(arow, s * 32, half),
                      b_frag_f32(wrow, s * 32, half), d);
#pragma unroll
    for (int r = 0; r < 8; ++r)
        out[((size_t)(b * N_) + i0 + half * 8 + r) * D_ + w * 16 + c] = d[r];
}

// =====================================================================
extern "C" void kernel_launch(void* const* d_in, const int* in_sizes, int n_in,
                              void* d_out, int out_size, void* d_ws, size_t ws_size,
                              hipStream_t stream) {
    const float* h_nodes = (const float*)d_in[0];
    const float* h_edges = (const float*)d_in[1];
    const float* Wq      = (const float*)d_in[2];
    const float* Wk      = (const float*)d_in[3];
    const float* Wv      = (const float*)d_in[4];
    const float* Wo      = (const float*)d_in[5];
    const float* Web     = (const float*)d_in[6];
    const float* Weg     = (const float*)d_in[7];

    char* ws = (char*)d_ws;
    float*  q_ws = (float*)(ws);                               // 1 MiB
    __bf16* k_ws = (__bf16*)(ws + (1u << 20));                 // 512 KiB
    float*  v_ws = (float*)(ws + (1u << 20) + (1u << 19));     // 1 MiB
    float*  a_ws = (float*)(ws + (2u << 20) + (1u << 19));     // 1 MiB

    dim3 gtile(N_ / 16, B_);   // 32 x 4
    dim3 grow(N_, B_);         // 512 x 4

    qkv_kernel<<<gtile, 256, 0, stream>>>(h_nodes, Wq, Wk, Wv, q_ws, k_ws, v_ws);
    attn_kernel<<<grow, 256, 0, stream>>>(h_edges, Web, Weg, q_ws, k_ws, v_ws, a_ws);
    out_kernel<<<gtile, 256, 0, stream>>>(a_ws, Wo, (float*)d_out);
}
